// MultiHeadedAttention_51994874085893
// MI455X (gfx1250) — matrix operand order repaired and checked by reading, not compiled, not run on hardware
//
#include <hip/hip_runtime.h>
#include <hip/hip_bf16.h>

#define H_     16
#define DMODEL 1024
#define DK     64
#define B_     4
#define S_     2048
#define NEGINF (-1.0e9f)

typedef __attribute__((ext_vector_type(16))) __bf16 v16bf;
typedef __attribute__((ext_vector_type(8)))  float  v8f;
typedef __attribute__((ext_vector_type(4)))  int    v4i_;

// addrspace-qualified int4 pointers matching the async-to-LDS builtin params
typedef __attribute__((address_space(1))) v4i_* as1_v4i;
typedef __attribute__((address_space(3))) v4i_* as3_v4i;

union Frag  { unsigned int u[8]; v16bf v; };
union Pack8 { uint4 u4; __bf16 h[8]; };

#if defined(__has_builtin)
#if __has_builtin(__builtin_amdgcn_global_load_async_to_lds_b128)
#define HAS_ASYNC_LDS 1
#endif
#if __has_builtin(__builtin_amdgcn_s_wait_asynccnt)
#define HAS_WAIT_ASYNC 1
#endif
#endif

// 16-byte memory -> LDS copy: async on CDNA5 (ASYNCcnt), fallback = via VGPRs.
__device__ __forceinline__ void copy16_to_lds(const __bf16* g, __bf16* l) {
#ifdef HAS_ASYNC_LDS
  __builtin_amdgcn_global_load_async_to_lds_b128(
      (as1_v4i)(void*)g, (as3_v4i)(void*)l, 0, 0);
#else
  *(uint4*)l = *(const uint4*)g;
#endif
}
__device__ __forceinline__ void copy_wait() {
#ifdef HAS_ASYNC_LDS
#ifdef HAS_WAIT_ASYNC
  __builtin_amdgcn_s_wait_asynccnt(0);
#else
  asm volatile("s_wait_asynccnt 0x0" ::: "memory");
#endif
#endif
}

__device__ __forceinline__ float row16_max(float x) {
#pragma unroll
  for (int off = 8; off; off >>= 1) x = fmaxf(x, __shfl_xor(x, off, 16));
  return x;
}
__device__ __forceinline__ float row16_sum(float x) {
#pragma unroll
  for (int off = 8; off; off >>= 1) x += __shfl_xor(x, off, 16);
  return x;
}

// ---------------------------------------------------------------------------
// GEMM: Y[m,n] = sum_k A[m,k] * W[n,k] + bias[n]
// M=8192, N=1024, K=1024.  Block tile 128x64, BLK_K=32, 8 waves (4x2),
// each wave computes 2x2 WMMA 16x16 tiles with v_wmma_f32_16x16x32_bf16.
// OUT_MODE 0: bf16 remapped to (b,h,s,d)       (Q, K projections)
// OUT_MODE 1: f32 row-major (m,n)              (final projection -> d_out)
// OUT_MODE 2: bf16 remapped TRANSPOSED (b,h,d,s) (V projection; makes the
//             attention PV B-matrix tile a contiguous copy -> async-able)
// ---------------------------------------------------------------------------
template <bool A_IS_BF16, int OUT_MODE>
__global__ __launch_bounds__(256)
void gemm_bf16_wmma(const void* __restrict__ Aptr, const float* __restrict__ W,
                    const float* __restrict__ bias, void* __restrict__ outp) {
  __shared__ __bf16 As[128][40];   // 80B rows: 16B-aligned, conflict-free frags
  __shared__ __bf16 Bs[64][40];
  const int tid  = threadIdx.x;
  const int lane = tid & 31, wave = tid >> 5;
  const int half = lane >> 4, ln = lane & 15;
  const int wm = wave >> 1, wn = wave & 1;
  const int m0 = blockIdx.x * 128;
  const int n0 = blockIdx.y * 64;

  v8f acc[2][2] = {};

  for (int k0 = 0; k0 < DMODEL; k0 += 32) {
    __syncthreads();
    // --- stage A tile ---
    if (!A_IS_BF16) {
      const float* A = (const float*)Aptr;
#pragma unroll
      for (int i = 0; i < 4; ++i) {
        int idx = tid + i * 256;            // 1024 float4s
        int r = idx >> 3, c = (idx & 7) << 2;
        float4 f = *(const float4*)(A + (size_t)(m0 + r) * DMODEL + k0 + c);
        As[r][c + 0] = (__bf16)f.x; As[r][c + 1] = (__bf16)f.y;
        As[r][c + 2] = (__bf16)f.z; As[r][c + 3] = (__bf16)f.w;
      }
    } else {
      const __bf16* A = (const __bf16*)Aptr;
#pragma unroll
      for (int i = 0; i < 2; ++i) {
        int idx = tid + i * 256;            // 512 x 16B chunks
        int r = idx >> 2, c = (idx & 3) << 3;
        copy16_to_lds(A + (size_t)(m0 + r) * DMODEL + k0 + c, &As[r][c]);
      }
    }
    // --- stage W tile (f32 -> bf16 conversion, stays synchronous) ---
#pragma unroll
    for (int i = 0; i < 2; ++i) {
      int idx = tid + i * 256;
      int r = idx >> 3, c = (idx & 7) << 2;
      float4 f = *(const float4*)(W + (size_t)(n0 + r) * DMODEL + k0 + c);
      Bs[r][c + 0] = (__bf16)f.x; Bs[r][c + 1] = (__bf16)f.y;
      Bs[r][c + 2] = (__bf16)f.z; Bs[r][c + 3] = (__bf16)f.w;
    }
    if (A_IS_BF16) copy_wait();
    __syncthreads();

    // --- fragments + WMMAs ---
    Frag a[2], b[2];
#pragma unroll
    for (int i = 0; i < 2; ++i) {
      int row = wm * 32 + i * 16 + ln;      // A: M = lane%16
#pragma unroll
      for (int j = 0; j < 8; ++j) {         // ISA 16-bit A 16x32 layout
        int kk = (j < 4 ? 2 * j : 16 + 2 * (j - 4)) + 8 * half;
        a[i].u[j] = *(const unsigned int*)&As[row][kk];
      }
    }
#pragma unroll
    for (int j = 0; j < 2; ++j) {
      int col = wn * 32 + j * 16 + ln;      // B: N = lane%16
#pragma unroll
      for (int t = 0; t < 8; ++t) {         // B 32x16: same K layout as A
        int kk = (t < 4 ? 2 * t : 16 + 2 * (t - 4)) + 8 * half;
        b[j].u[t] = *(const unsigned int*)&Bs[col][kk];
      }
    }
#pragma unroll
    for (int i = 0; i < 2; ++i)
#pragma unroll
      for (int j = 0; j < 2; ++j)
        acc[i][j] = __builtin_amdgcn_wmma_f32_16x16x32_bf16(
            false, a[i].v, false, b[j].v, (short)0, acc[i][j], false, false);
  }

  // --- epilogue: + bias, store ---
#pragma unroll
  for (int j = 0; j < 2; ++j) {
    int n = n0 + wn * 32 + j * 16 + ln;
    float bv = bias[n];
#pragma unroll
    for (int i = 0; i < 2; ++i) {
#pragma unroll
      for (int r = 0; r < 8; ++r) {
        int m = m0 + wm * 32 + i * 16 + r + 8 * half;  // C layout: M = r + 8*half
        float y = acc[i][j][r] + bv;
        int bb = m >> 11, s = m & (S_ - 1);
        int hh = n >> 6,  d = n & (DK - 1);
        if (OUT_MODE == 0) {
          ((__bf16*)outp)[(((size_t)(bb * H_ + hh)) * S_ + s) * DK + d] = (__bf16)y;
        } else if (OUT_MODE == 2) {
          ((__bf16*)outp)[(((size_t)(bb * H_ + hh)) * DK + d) * S_ + s] = (__bf16)y;
        } else {
          ((float*)outp)[(size_t)m * DMODEL + n] = y;
        }
      }
    }
  }
}

// ---------------------------------------------------------------------------
// Fused flash attention.  Grid = (S/128, B*H).  Block = 256 (8 waves).
// qp,kp: (b,h,s,d) bf16.  vt: (b,h,d,s) bf16 (pre-transposed by projection).
// All tile staging is contiguous 16B chunks -> GLOBAL_LOAD_ASYNC_TO_LDS_B128.
// ---------------------------------------------------------------------------
__global__ __launch_bounds__(256)
void attn_wmma(const unsigned short* __restrict__ qp_,
               const unsigned short* __restrict__ kp_,
               const unsigned short* __restrict__ vt_,
               const int* __restrict__ mask,
               unsigned short* __restrict__ outp_) {
  const __bf16* qp = (const __bf16*)qp_;
  const __bf16* kp = (const __bf16*)kp_;
  const __bf16* vt = (const __bf16*)vt_;
  __bf16* outp = (__bf16*)outp_;

  __shared__ __bf16 Qs[128][72];     // (q_local, d); 144B rows (16B aligned)
  __shared__ __bf16 Ks[64][72];      // (kv_local, d)
  __shared__ __bf16 Vt[64][72];      // (d, kv_local) -- already transposed in ws
  __shared__ __bf16 Ps[8][16][72];   // per-wave P tile (m, kv_local)

  const int tid  = threadIdx.x;
  const int lane = tid & 31, wave = tid >> 5;
  const int half = lane >> 4, ln = lane & 15;
  const int bh = blockIdx.y;
  const int b  = bh >> 4;
  const int h  = bh & 15;
  const int q0 = blockIdx.x * 128;
  const float scale = 0.125f;        // 1/sqrt(64)

  // stage Q tile (resident for whole kernel): 128x64 bf16 = 1024 x 16B
  const __bf16* qbase = qp + ((size_t)bh * S_ + q0) * DK;
#pragma unroll
  for (int i = 0; i < 4; ++i) {
    int idx = tid + i * 256;
    int r = idx >> 3, c = (idx & 7) << 3;
    copy16_to_lds(qbase + r * DK + c, &Qs[r][c]);
  }

  v8f oacc[4] = {};
  float rmax[8], rsum[8];
#pragma unroll
  for (int r = 0; r < 8; ++r) { rmax[r] = NEGINF; rsum[r] = 0.f; }

  for (int kv0 = 0; kv0 < S_; kv0 += 64) {
    __syncthreads();                  // previous tile fully consumed
    const __bf16* kbase  = kp + ((size_t)bh * S_ + kv0) * DK;
    const __bf16* vtbase = vt + ((size_t)bh * DK) * S_ + kv0;
#pragma unroll
    for (int i = 0; i < 2; ++i) {
      int idx = tid + i * 256;        // 512 x 16B per tile
      int r = idx >> 3, c = (idx & 7) << 3;
      copy16_to_lds(kbase + r * DK + c, &Ks[r][c]);          // (kv, d)
      copy16_to_lds(vtbase + (size_t)r * S_ + c, &Vt[r][c]); // (d, kv)
    }
    copy_wait();
    __syncthreads();

    // S = Q(16x64) @ K^T(64x64): 2 k-steps x 4 n-tiles
    v8f sacc[4] = {};
#pragma unroll
    for (int ks = 0; ks < 2; ++ks) {
      Frag a;
      int row = wave * 16 + ln;
#pragma unroll
      for (int j = 0; j < 8; ++j) {
        int kk = ks * 32 + (j < 4 ? 2 * j : 16 + 2 * (j - 4)) + 8 * half;
        a.u[j] = *(const unsigned int*)&Qs[row][kk];
      }
#pragma unroll
      for (int j = 0; j < 4; ++j) {
        Frag bb;
        int col = j * 16 + ln;
#pragma unroll
        for (int t = 0; t < 8; ++t) {
          int kk = ks * 32 + (t < 4 ? 2 * t : 16 + 2 * (t - 4)) + 8 * half;
          bb.u[t] = *(const unsigned int*)&Ks[col][kk];
        }
        sacc[j] = __builtin_amdgcn_wmma_f32_16x16x32_bf16(
            false, a.v, false, bb.v, (short)0, sacc[j], false, false);
      }
    }

    // mask + online softmax update (one row == one VGPR over a 16-lane half)
#pragma unroll
    for (int r = 0; r < 8; ++r) {
      int qrow = q0 + wave * 16 + r + 8 * half;
      const int* mrow = mask + ((size_t)b * S_ + qrow) * S_ + kv0;
      float mx = NEGINF;
#pragma unroll
      for (int j = 0; j < 4; ++j) {
        int mk = mrow[j * 16 + ln];
        float sv = mk ? sacc[j][r] * scale : NEGINF;
        sacc[j][r] = sv;
        mx = fmaxf(mx, sv);
      }
      mx = row16_max(mx);
      float mnew  = fmaxf(rmax[r], mx);
      float alpha = __expf(rmax[r] - mnew);
      rmax[r] = mnew;
      float psum = 0.f;
#pragma unroll
      for (int j = 0; j < 4; ++j) {
        float pv = __expf(sacc[j][r] - mnew);   // masked -> exp(-1e9-m) == 0 in f32
        psum += pv;
        Ps[wave][r + 8 * half][j * 16 + ln] = (__bf16)pv;
      }
      rsum[r] = rsum[r] * alpha + row16_sum(psum);
#pragma unroll
      for (int j = 0; j < 4; ++j) oacc[j][r] *= alpha;
    }

    // O += P(16x64) @ V(64x64): P re-read from LDS in A-fragment layout
#pragma unroll
    for (int ks = 0; ks < 2; ++ks) {
      Frag a;
#pragma unroll
      for (int j = 0; j < 8; ++j) {
        int kk = ks * 32 + (j < 4 ? 2 * j : 16 + 2 * (j - 4)) + 8 * half;
        a.u[j] = *(const unsigned int*)&Ps[wave][ln][kk];
      }
#pragma unroll
      for (int j = 0; j < 4; ++j) {
        Frag bb;
        int col = j * 16 + ln;                   // N = d index
#pragma unroll
        for (int t = 0; t < 8; ++t) {
          int kk = ks * 32 + (t < 4 ? 2 * t : 16 + 2 * (t - 4)) + 8 * half;  // K = kv index (contig in Vt)
          bb.u[t] = *(const unsigned int*)&Vt[col][kk];
        }
        oacc[j] = __builtin_amdgcn_wmma_f32_16x16x32_bf16(
            false, a.v, false, bb.v, (short)0, oacc[j], false, false);
      }
    }
  }

  // epilogue: normalize, store as (b, s, h*64 + d) bf16
#pragma unroll
  for (int r = 0; r < 8; ++r) {
    int qrow = q0 + wave * 16 + r + 8 * half;
    float inv = rsum[r] > 0.f ? 1.0f / rsum[r] : 0.0f;
#pragma unroll
    for (int j = 0; j < 4; ++j) {
      float y = oacc[j][r] * inv;
      outp[((size_t)b * S_ + qrow) * DMODEL + h * DK + j * 16 + ln] = (__bf16)y;
    }
  }
}

// ---------------------------------------------------------------------------
extern "C" void kernel_launch(void* const* d_in, const int* in_sizes, int n_in,
                              void* d_out, int out_size, void* d_ws, size_t ws_size,
                              hipStream_t stream) {
  const float* query = (const float*)d_in[0];
  const float* key_  = (const float*)d_in[1];
  const float* value = (const float*)d_in[2];
  const int*   mask  = (const int*)d_in[3];
  const float* Wq = (const float*)d_in[4];
  const float* bq = (const float*)d_in[5];
  const float* Wk = (const float*)d_in[6];
  const float* bk = (const float*)d_in[7];
  const float* Wv = (const float*)d_in[8];
  const float* bv = (const float*)d_in[9];
  const float* Wo = (const float*)d_in[10];
  const float* bo = (const float*)d_in[11];

  const size_t elems = (size_t)B_ * S_ * DMODEL;     // 8M elements
  unsigned short* qp = (unsigned short*)d_ws;        // bf16 buffers in ws
  unsigned short* kp = qp + elems;
  unsigned short* vt = kp + elems;                   // (b,h,d,s) transposed
  unsigned short* ao = vt + elems;                   // total 64 MB

  dim3 blk(256);
  dim3 ggrid(8192 / 128, DMODEL / 64);   // (64, 16)
  gemm_bf16_wmma<false, 0><<<ggrid, blk, 0, stream>>>(query, Wq, bq, (void*)qp);
  gemm_bf16_wmma<false, 0><<<ggrid, blk, 0, stream>>>(key_,  Wk, bk, (void*)kp);
  gemm_bf16_wmma<false, 2><<<ggrid, blk, 0, stream>>>(value, Wv, bv, (void*)vt);

  dim3 agrid(S_ / 128, B_ * H_);         // (16, 64)
  attn_wmma<<<agrid, blk, 0, stream>>>(qp, kp, vt, mask, ao);

  gemm_bf16_wmma<true, 1><<<ggrid, blk, 0, stream>>>((const void*)ao, Wo, bo, d_out);
}
